// TransformerSummarizer_8443905704420
// MI455X (gfx1250) — matrix operand order repaired and checked by reading, not compiled, not run on hardware
//
#include <hip/hip_runtime.h>
#include <math.h>

// ---------------- model constants ----------------
#define DMODEL 512
#define DHEAD  64
#define NHEAD  8
#define SEQ_E  1024
#define SEQ_D  256
#define BATCH  4
#define KSEL   32

typedef __attribute__((ext_vector_type(16))) __bf16 v16bf;
typedef __attribute__((ext_vector_type(8)))  __bf16 v8bf;
typedef __attribute__((ext_vector_type(8)))  float  v8f;
typedef int i32x4 __attribute__((vector_size(4 * sizeof(int))));

// ---- optional CDNA5 async global->LDS path (ASYNCcnt), with safe fallback ----
#if defined(__has_builtin)
#if __has_builtin(__builtin_amdgcn_global_load_async_to_lds_b128) && \
    __has_builtin(__builtin_amdgcn_s_wait_asynccnt)
#define ASYNC_LDS 1
#endif
#endif

#ifdef ASYNC_LDS
// Prototype (from clang diagnostic): (int4 AS1* gsrc, int4 AS3* lds, imm off, imm cpol)
__device__ __forceinline__ void async_cp16(const __bf16* g, __bf16* l)
{
    __builtin_amdgcn_global_load_async_to_lds_b128(
        (__attribute__((address_space(1))) i32x4*)(unsigned long long)(g),
        (__attribute__((address_space(3))) i32x4*)(unsigned long long)(l),
        0, 0);
}
#endif

// =====================================================================
// f32 -> bf16 bulk convert (8 elems/thread, vector stores). n % 2048 == 0.
// =====================================================================
__global__ __launch_bounds__(256)
void k_cvt_bf16(const float* __restrict__ in, __bf16* __restrict__ out)
{
    const size_t i = ((size_t)blockIdx.x * 256 + threadIdx.x) * 8;
    v8bf o;
    #pragma unroll
    for (int j = 0; j < 8; ++j) o[j] = (__bf16)in[i + j];
    *(v8bf*)(out + i) = o;
}

// =====================================================================
// bf16 GEMM: C[M,N] = act(A[M,K] @ W[N,K]^T + bias) + residual
//   A, W bf16; accumulate f32 via V_WMMA_F32_16X16X32_BF16.
//   Block = 256 thr = 8 waves (2 along M x 4 along N); block tile 64x256;
//   wave tile 32x64 as 2x4 fragments; K-step 32 staged in LDS
//   (async global->LDS when available, else ds stores).
//   Output: exactly one of Cf (f32) / Cb (bf16) non-null.
//   Requires M%64==0, N%256==0, K%32==0 (true for all shapes here).
// =====================================================================
__global__ __launch_bounds__(256)
void k_gemm_bf16(const __bf16* __restrict__ A, const __bf16* __restrict__ W,
                 const float* __restrict__ bias, const float* __restrict__ residual,
                 float* __restrict__ Cf, __bf16* __restrict__ Cb,
                 int M, int N, int K, int act_gelu)
{
    __shared__ __bf16 As[64][32];    //  4 KB
    __shared__ __bf16 Bs[256][32];   // 16 KB
    const int tid   = threadIdx.x;
    const int lane  = tid & 31;
    const int wid   = tid >> 5;
    const int waveM = wid & 1;
    const int waveN = wid >> 1;
    const int bM = blockIdx.y * 64;
    const int bN = blockIdx.x * 256;
    const int r16  = lane & 15;
    const int half = lane >> 4;

    // staging assignments: A tile 64 rows x 32 (1 x 16B chunk / thread),
    //                      B tile 256 rows x 32 (4 x 16B chunks / thread)
    const int arow = tid >> 2;
    const int acol = (tid & 3) * 8;
    const __bf16* aG = A + (size_t)(bM + arow) * K + acol;
    const __bf16* wG = W + (size_t)(bN + tid) * K;

    v8f acc[2][4] = {};

    for (int k0 = 0; k0 < K; k0 += 32) {
        __builtin_prefetch(wG + k0 + 32, 0, 1);   // next K-step, speculative
#ifdef ASYNC_LDS
        async_cp16(aG + k0, &As[arow][acol]);
        #pragma unroll
        for (int c = 0; c < 4; ++c)
            async_cp16(wG + k0 + 8 * c, &Bs[tid][8 * c]);
        __builtin_amdgcn_s_wait_asynccnt(0);
        __syncthreads();
#else
        *(v8bf*)&As[arow][acol] = *(const v8bf*)(aG + k0);
        #pragma unroll
        for (int c = 0; c < 4; ++c)
            *(v8bf*)&Bs[tid][8 * c] = *(const v8bf*)(wG + k0 + 8 * c);
        __syncthreads();
#endif
        // ---- fragment reads from LDS (ds_load_b128) ----
        v16bf af[2], bf[4];
        #pragma unroll
        for (int mi = 0; mi < 2; ++mi) {
            // A layout: elems 0..7 -> K = 8*half + j ; elems 8..15 -> +16
            const v8bf lo = *(const v8bf*)&As[waveM * 32 + mi * 16 + r16][8 * half];
            const v8bf hi = *(const v8bf*)&As[waveM * 32 + mi * 16 + r16][8 * half + 16];
            af[mi] = __builtin_shufflevector(lo, hi, 0, 1, 2, 3, 4, 5, 6, 7,
                                                     8, 9, 10, 11, 12, 13, 14, 15);
        }
        #pragma unroll
        for (int ni = 0; ni < 4; ++ni) {
            // B layout: lane -> column, elems 0..7 -> K = 8*half + j ; elems 8..15 -> +16
            const v8bf lo = *(const v8bf*)&Bs[waveN * 64 + ni * 16 + r16][8 * half];
            const v8bf hi = *(const v8bf*)&Bs[waveN * 64 + ni * 16 + r16][8 * half + 16];
            bf[ni] = __builtin_shufflevector(lo, hi, 0, 1, 2, 3, 4, 5, 6, 7,
                                                     8, 9, 10, 11, 12, 13, 14, 15);
        }
        #pragma unroll
        for (int mi = 0; mi < 2; ++mi)
            #pragma unroll
            for (int ni = 0; ni < 4; ++ni)
                acc[mi][ni] = __builtin_amdgcn_wmma_f32_16x16x32_bf16(
                    false, af[mi], false, bf[ni], (short)0, acc[mi][ni], false, false);
        __syncthreads();
    }

    // ---- epilogue: C/D layout: vgpr r -> m = r + 8*half, lane%16 -> n ----
    #pragma unroll
    for (int mi = 0; mi < 2; ++mi) {
        #pragma unroll
        for (int ni = 0; ni < 4; ++ni) {
            #pragma unroll
            for (int r = 0; r < 8; ++r) {
                const int m = bM + waveM * 32 + mi * 16 + r + 8 * half;
                const int n = bN + waveN * 64 + ni * 16 + r16;
                float v = acc[mi][ni][r];
                if (bias)     v += bias[n];
                if (act_gelu) v = 0.5f * v * (1.0f + erff(v * 0.70710678f));
                if (residual) v += residual[(size_t)m * N + n];
                if (Cb) Cb[(size_t)m * N + n] = (__bf16)v;
                else    Cf[(size_t)m * N + n] = v;
            }
        }
    }
}

// =====================================================================
// Embedding gather: out[b,t,:] = tok_emb[ids[b,t]] + pos[t]  (f32)
// =====================================================================
__global__ __launch_bounds__(256)
void k_embed(const int* __restrict__ ids, const float* __restrict__ emb,
             const float* __restrict__ pos, float* __restrict__ out, int T)
{
    const size_t i  = (size_t)blockIdx.x * blockDim.x + threadIdx.x;
    const int d     = (int)(i & (DMODEL - 1));
    const size_t tk = i >> 9;
    const int t     = (int)(tk % T);
    const int id    = ids[tk];
    out[i] = emb[(size_t)id * DMODEL + d] + pos[(size_t)t * DMODEL + d];
}

// =====================================================================
// LayerNorm over D=512, one wave per row, bf16 output (feeds GEMM A)
// =====================================================================
__global__ __launch_bounds__(256)
void k_layernorm(const float* __restrict__ x, const float* __restrict__ g,
                 const float* __restrict__ b, __bf16* __restrict__ out, int rows)
{
    const int wrow = (int)((blockIdx.x * (size_t)blockDim.x + threadIdx.x) >> 5);
    const int lane = threadIdx.x & 31;
    if (wrow >= rows) return;
    const float* xr = x + (size_t)wrow * DMODEL;
    float v[16], s = 0.f;
    #pragma unroll
    for (int j = 0; j < 16; ++j) { v[j] = xr[lane + 32 * j]; s += v[j]; }
    #pragma unroll
    for (int off = 16; off; off >>= 1) s += __shfl_xor(s, off, 32);
    const float mu = s * (1.f / DMODEL);
    float var = 0.f;
    #pragma unroll
    for (int j = 0; j < 16; ++j) { const float d = v[j] - mu; var += d * d; }
    #pragma unroll
    for (int off = 16; off; off >>= 1) var += __shfl_xor(var, off, 32);
    const float rs = rsqrtf(var * (1.f / DMODEL) + 1e-5f);
    __bf16* o = out + (size_t)wrow * DMODEL;
    #pragma unroll
    for (int j = 0; j < 16; ++j) {
        const int d = lane + 32 * j;
        o[d] = (__bf16)(g[d] * (v[j] - mu) * rs + b[d]);
    }
}

// =====================================================================
// Encoder sparse attention (top-32): one wave per (b,h,t), bf16 out
// =====================================================================
__global__ __launch_bounds__(256)
void k_attn_topk(const float* __restrict__ qkv, const int* __restrict__ mask,
                 __bf16* __restrict__ out)
{
    const int gw   = (int)((blockIdx.x * (size_t)blockDim.x + threadIdx.x) >> 5);
    const int lane = threadIdx.x & 31;
    const int lw   = threadIdx.x >> 5;
    const int t  = gw % SEQ_E;
    const int hh = (gw / SEQ_E) & (NHEAD - 1);
    const int bb = gw / (SEQ_E * NHEAD);

    const float* qrow  = qkv + (size_t)(bb * SEQ_E + t) * 3 * DMODEL + hh * DHEAD;
    const float* kbase = qkv + (size_t)(bb * SEQ_E) * 3 * DMODEL + DMODEL + hh * DHEAD;
    const float* vbase = qkv + (size_t)(bb * SEQ_E) * 3 * DMODEL + 2 * DMODEL + hh * DHEAD;
    const int*   mrow  = mask + bb * SEQ_E;

    __shared__ float qs[8][DHEAD];
    qs[lw][lane]      = qrow[lane];
    qs[lw][lane + 32] = qrow[lane + 32];
    __syncthreads();

    float sc[32];
    #pragma unroll 4
    for (int j = 0; j < 32; ++j) {
        const int s = lane + 32 * j;
        const float* kr = kbase + (size_t)s * 3 * DMODEL;
        float acc = 0.f;
        for (int d = 0; d < DHEAD; ++d) acc += qs[lw][d] * kr[d];
        sc[j] = (mrow[s] == 0) ? -3e38f : acc * 0.125f;   // 1/sqrt(64)
    }

    unsigned used = 0;
    float m0 = 0.f, denom = 0.f, o0 = 0.f, o1 = 0.f;
    for (int it = 0; it < KSEL; ++it) {
        float bv = -3.4e38f; int bj = 0;
        #pragma unroll
        for (int j = 0; j < 32; ++j)
            if (!((used >> j) & 1u) && sc[j] > bv) { bv = sc[j]; bj = j; }
        int bs = bj * 32 + lane;
        #pragma unroll
        for (int off = 16; off; off >>= 1) {
            const float ov = __shfl_xor(bv, off, 32);
            const int   os = __shfl_xor(bs, off, 32);
            if (ov > bv || (ov == bv && os < bs)) { bv = ov; bs = os; }
        }
        if ((bs & 31) == lane) used |= 1u << (bs >> 5);
        if (it == 0) m0 = bv;
        const float w = __expf(bv - m0);
        denom += w;
        const float* vr = vbase + (size_t)bs * 3 * DMODEL;
        o0 += w * vr[lane];
        o1 += w * vr[lane + 32];
    }
    const float inv = 1.f / denom;
    __bf16* orow = out + (size_t)(bb * SEQ_E + t) * DMODEL + hh * DHEAD;
    orow[lane]      = (__bf16)(o0 * inv);
    orow[lane + 32] = (__bf16)(o1 * inv);
}

// =====================================================================
// Decoder causal self-attention: one wave per (b,h,t), 256 keys, bf16 out
// =====================================================================
__global__ __launch_bounds__(256)
void k_attn_causal(const float* __restrict__ qkv, __bf16* __restrict__ out)
{
    const int gw   = (int)((blockIdx.x * (size_t)blockDim.x + threadIdx.x) >> 5);
    const int lane = threadIdx.x & 31;
    const int lw   = threadIdx.x >> 5;
    const int t  = gw % SEQ_D;
    const int hh = (gw / SEQ_D) & (NHEAD - 1);
    const int bb = gw / (SEQ_D * NHEAD);

    const float* qrow  = qkv + (size_t)(bb * SEQ_D + t) * 3 * DMODEL + hh * DHEAD;
    const float* kbase = qkv + (size_t)(bb * SEQ_D) * 3 * DMODEL + DMODEL + hh * DHEAD;
    const float* vbase = qkv + (size_t)(bb * SEQ_D) * 3 * DMODEL + 2 * DMODEL + hh * DHEAD;

    __shared__ float qs[8][DHEAD];
    qs[lw][lane]      = qrow[lane];
    qs[lw][lane + 32] = qrow[lane + 32];
    __syncthreads();

    float sc[8];
    #pragma unroll
    for (int j = 0; j < 8; ++j) {
        const int s = lane + 32 * j;
        float acc = -3e38f;
        if (s <= t) {
            const float* kr = kbase + (size_t)s * 3 * DMODEL;
            acc = 0.f;
            for (int d = 0; d < DHEAD; ++d) acc += qs[lw][d] * kr[d];
            acc *= 0.125f;
        }
        sc[j] = acc;
    }
    float mx = -3e38f;
    #pragma unroll
    for (int j = 0; j < 8; ++j) mx = fmaxf(mx, sc[j]);
    #pragma unroll
    for (int off = 16; off; off >>= 1) mx = fmaxf(mx, __shfl_xor(mx, off, 32));
    float p[8], sum = 0.f;
    #pragma unroll
    for (int j = 0; j < 8; ++j) { p[j] = __expf(sc[j] - mx); sum += p[j]; }
    #pragma unroll
    for (int off = 16; off; off >>= 1) sum += __shfl_xor(sum, off, 32);
    const float inv = 1.f / sum;

    float o0 = 0.f, o1 = 0.f;
    for (int j = 0; j < 8; ++j) {
        for (int l = 0; l < 32; ++l) {
            const int s = 32 * j + l;
            if (s > t) break;                       // t uniform across wave
            const float w = __shfl(p[j], l, 32);
            const float* vr = vbase + (size_t)s * 3 * DMODEL;
            o0 += w * vr[lane];
            o1 += w * vr[lane + 32];
        }
    }
    __bf16* orow = out + (size_t)(bb * SEQ_D + t) * DMODEL + hh * DHEAD;
    orow[lane]      = (__bf16)(o0 * inv);
    orow[lane + 32] = (__bf16)(o1 * inv);
}

// =====================================================================
// Cross attention: full softmax over 1024 masked keys, bf16 out
// =====================================================================
__global__ __launch_bounds__(256)
void k_attn_cross(const float* __restrict__ q, const float* __restrict__ kv,
                  const int* __restrict__ mask, __bf16* __restrict__ out)
{
    const int gw   = (int)((blockIdx.x * (size_t)blockDim.x + threadIdx.x) >> 5);
    const int lane = threadIdx.x & 31;
    const int lw   = threadIdx.x >> 5;
    const int t  = gw % SEQ_D;
    const int hh = (gw / SEQ_D) & (NHEAD - 1);
    const int bb = gw / (SEQ_D * NHEAD);

    const float* qrow  = q  + (size_t)(bb * SEQ_D + t) * DMODEL + hh * DHEAD;
    const float* kbase = kv + (size_t)(bb * SEQ_E) * 2 * DMODEL + hh * DHEAD;
    const float* vbase = kv + (size_t)(bb * SEQ_E) * 2 * DMODEL + DMODEL + hh * DHEAD;
    const int*   mrow  = mask + bb * SEQ_E;

    __shared__ float qs[8][DHEAD];
    qs[lw][lane]      = qrow[lane];
    qs[lw][lane + 32] = qrow[lane + 32];
    __syncthreads();

    float sc[32];
    #pragma unroll 4
    for (int j = 0; j < 32; ++j) {
        const int s = lane + 32 * j;
        const float* kr = kbase + (size_t)s * 2 * DMODEL;
        float acc = 0.f;
        for (int d = 0; d < DHEAD; ++d) acc += qs[lw][d] * kr[d];
        sc[j] = (mrow[s] == 0) ? -3e38f : acc * 0.125f;
    }
    float mx = -3e38f;
    #pragma unroll
    for (int j = 0; j < 32; ++j) mx = fmaxf(mx, sc[j]);
    #pragma unroll
    for (int off = 16; off; off >>= 1) mx = fmaxf(mx, __shfl_xor(mx, off, 32));
    float p[32], sum = 0.f;
    #pragma unroll
    for (int j = 0; j < 32; ++j) { p[j] = __expf(sc[j] - mx); sum += p[j]; }
    #pragma unroll
    for (int off = 16; off; off >>= 1) sum += __shfl_xor(sum, off, 32);
    const float inv = 1.f / sum;

    float o0 = 0.f, o1 = 0.f;
    for (int j = 0; j < 32; ++j) {
        for (int l = 0; l < 32; ++l) {
            const float w = __shfl(p[j], l, 32);
            const int s = 32 * j + l;
            const float* vr = vbase + (size_t)s * 2 * DMODEL;
            o0 += w * vr[lane];
            o1 += w * vr[lane + 32];
        }
    }
    __bf16* orow = out + (size_t)(bb * SEQ_D + t) * DMODEL + hh * DHEAD;
    orow[lane]      = (__bf16)(o0 * inv);
    orow[lane + 32] = (__bf16)(o1 * inv);
}

// =====================================================================
// Host-side orchestration
// =====================================================================
extern "C" void kernel_launch(void* const* d_in, const int* in_sizes, int n_in,
                              void* d_out, int out_size, void* d_ws, size_t ws_size,
                              hipStream_t stream)
{
    (void)in_sizes; (void)n_in; (void)out_size; (void)ws_size;
    const int*   input_ids = (const int*)  d_in[0];
    const int*   attn_mask = (const int*)  d_in[1];
    const int*   dec_ids   = (const int*)  d_in[2];
    const float* tok_emb   = (const float*)d_in[3];
    const float* pos_enc   = (const float*)d_in[4];
    const float* pos_dec   = (const float*)d_in[5];
    const float* enc_ln1_g = (const float*)d_in[6];
    const float* enc_ln1_b = (const float*)d_in[7];
    const float* enc_qkv_w = (const float*)d_in[8];
    const float* enc_qkv_b = (const float*)d_in[9];
    const float* enc_out_w = (const float*)d_in[10];
    const float* enc_out_b = (const float*)d_in[11];
    const float* enc_ln2_g = (const float*)d_in[12];
    const float* enc_ln2_b = (const float*)d_in[13];
    const float* enc_fc1_w = (const float*)d_in[14];
    const float* enc_fc1_b = (const float*)d_in[15];
    const float* enc_fc2_w = (const float*)d_in[16];
    const float* enc_fc2_b = (const float*)d_in[17];
    const float* dec_ln1_g = (const float*)d_in[18];
    const float* dec_ln1_b = (const float*)d_in[19];
    const float* dec_in_w  = (const float*)d_in[20];
    const float* dec_in_b  = (const float*)d_in[21];
    const float* dec_sao_w = (const float*)d_in[22];
    const float* dec_sao_b = (const float*)d_in[23];
    const float* dec_ln2_g = (const float*)d_in[24];
    const float* dec_ln2_b = (const float*)d_in[25];
    const float* dec_q_w   = (const float*)d_in[26];
    const float* dec_q_b   = (const float*)d_in[27];
    const float* dec_kv_w  = (const float*)d_in[28];
    const float* dec_kv_b  = (const float*)d_in[29];
    const float* dec_co_w  = (const float*)d_in[30];
    const float* dec_co_b  = (const float*)d_in[31];
    const float* dec_ln3_g = (const float*)d_in[32];
    const float* dec_ln3_b = (const float*)d_in[33];
    const float* dec_fc1_w = (const float*)d_in[34];
    const float* dec_fc1_b = (const float*)d_in[35];
    const float* dec_fc2_w = (const float*)d_in[36];
    const float* dec_fc2_b = (const float*)d_in[37];
    const float* ln_enc_g  = (const float*)d_in[38];
    const float* ln_enc_b  = (const float*)d_in[39];
    const float* ln_dec_g  = (const float*)d_in[40];
    const float* ln_dec_b  = (const float*)d_in[41];
    const float* lm_w      = (const float*)d_in[42];

    const int Me = BATCH * SEQ_E;   // 4096 tokens
    const int Md = BATCH * SEQ_D;   // 1024 tokens

    // ---- workspace carve-up: f32 region then bf16 region ----
    float* wsf = (float*)d_ws;
    size_t off = 0;
    auto allocf = [&](size_t n) { float* p = wsf + off; off += n; return p; };
    float* xe   = allocf((size_t)Me * DMODEL);        // encoder residual
    float* qkve = allocf((size_t)Me * 3 * DMODEL);
    float* qkvd = allocf((size_t)Md * 3 * DMODEL);
    float* xd   = allocf((size_t)Md * DMODEL);
    float* qc   = allocf((size_t)Md * DMODEL);
    float* kvc  = allocf((size_t)Me * 2 * DMODEL);

    __bf16* wsb = (__bf16*)(wsf + off);
    size_t offb = 0;
    auto allocb = [&](size_t n) { __bf16* p = wsb + offb; offb += n; return p; };
    __bf16* he_b   = allocb((size_t)Me * DMODEL);
    __bf16* aoe_b  = allocb((size_t)Me * DMODEL);
    __bf16* ffne_b = allocb((size_t)Me * 4 * DMODEL);
    __bf16* mem_b  = allocb((size_t)Me * DMODEL);
    __bf16* hd_b   = allocb((size_t)Md * DMODEL);
    __bf16* aod_b  = allocb((size_t)Md * DMODEL);
    __bf16* ffnd_b = allocb((size_t)Md * 4 * DMODEL);
    // bf16 weights (converted once per launch)
    __bf16* w_qkv_e = allocb(2UL * 3 * DMODEL * DMODEL);
    __bf16* w_out_e = allocb(2UL * DMODEL * DMODEL);
    __bf16* w_fc1_e = allocb(2UL * 4 * DMODEL * DMODEL);
    __bf16* w_fc2_e = allocb(2UL * DMODEL * 4 * DMODEL);
    __bf16* w_in_d  = allocb(2UL * 3 * DMODEL * DMODEL);
    __bf16* w_sao_d = allocb(2UL * DMODEL * DMODEL);
    __bf16* w_q_d   = allocb(2UL * DMODEL * DMODEL);
    __bf16* w_kv_d  = allocb(2UL * 2 * DMODEL * DMODEL);
    __bf16* w_co_d  = allocb(2UL * DMODEL * DMODEL);
    __bf16* w_fc1_d = allocb(2UL * 4 * DMODEL * DMODEL);
    __bf16* w_fc2_d = allocb(2UL * DMODEL * 4 * DMODEL);
    __bf16* w_lm    = allocb(32000UL * DMODEL);

    auto cvt = [&](const float* src, __bf16* dst, size_t n) {
        k_cvt_bf16<<<(unsigned)(n / 2048), 256, 0, stream>>>(src, dst);
    };
    cvt(enc_qkv_w, w_qkv_e, 2UL * 3 * DMODEL * DMODEL);
    cvt(enc_out_w, w_out_e, 2UL * DMODEL * DMODEL);
    cvt(enc_fc1_w, w_fc1_e, 2UL * 4 * DMODEL * DMODEL);
    cvt(enc_fc2_w, w_fc2_e, 2UL * DMODEL * 4 * DMODEL);
    cvt(dec_in_w,  w_in_d,  2UL * 3 * DMODEL * DMODEL);
    cvt(dec_sao_w, w_sao_d, 2UL * DMODEL * DMODEL);
    cvt(dec_q_w,   w_q_d,   2UL * DMODEL * DMODEL);
    cvt(dec_kv_w,  w_kv_d,  2UL * 2 * DMODEL * DMODEL);
    cvt(dec_co_w,  w_co_d,  2UL * DMODEL * DMODEL);
    cvt(dec_fc1_w, w_fc1_d, 2UL * 4 * DMODEL * DMODEL);
    cvt(dec_fc2_w, w_fc2_d, 2UL * DMODEL * 4 * DMODEL);
    cvt(lm_w,      w_lm,    32000UL * DMODEL);

    auto gemm = [&](const __bf16* A, const __bf16* W, const float* bias,
                    const float* res, float* Cf, __bf16* Cb,
                    int M, int N, int K, int act) {
        dim3 grid(N / 256, M / 64);
        k_gemm_bf16<<<grid, 256, 0, stream>>>(A, W, bias, res, Cf, Cb, M, N, K, act);
    };
    auto ln = [&](const float* x, const float* g, const float* b, __bf16* o, int rows) {
        k_layernorm<<<rows / 8, 256, 0, stream>>>(x, g, b, o, rows);
    };

    // ================= encoder =================
    k_embed<<<(Me * DMODEL) / 256, 256, 0, stream>>>(input_ids, tok_emb, pos_enc, xe, SEQ_E);
    for (int i = 0; i < 2; ++i) {
        ln(xe, enc_ln1_g + i * DMODEL, enc_ln1_b + i * DMODEL, he_b, Me);
        gemm(he_b, w_qkv_e + (size_t)i * 3 * DMODEL * DMODEL, enc_qkv_b + i * 3 * DMODEL,
             nullptr, qkve, nullptr, Me, 3 * DMODEL, DMODEL, 0);
        k_attn_topk<<<(BATCH * NHEAD * SEQ_E) / 8, 256, 0, stream>>>(qkve, attn_mask, aoe_b);
        gemm(aoe_b, w_out_e + (size_t)i * DMODEL * DMODEL, enc_out_b + i * DMODEL,
             xe, xe, nullptr, Me, DMODEL, DMODEL, 0);
        ln(xe, enc_ln2_g + i * DMODEL, enc_ln2_b + i * DMODEL, he_b, Me);
        gemm(he_b, w_fc1_e + (size_t)i * 4 * DMODEL * DMODEL, enc_fc1_b + i * 4 * DMODEL,
             nullptr, nullptr, ffne_b, Me, 4 * DMODEL, DMODEL, 1);
        gemm(ffne_b, w_fc2_e + (size_t)i * DMODEL * 4 * DMODEL, enc_fc2_b + i * DMODEL,
             xe, xe, nullptr, Me, DMODEL, 4 * DMODEL, 0);
    }
    ln(xe, ln_enc_g, ln_enc_b, mem_b, Me);

    // ================= decoder =================
    k_embed<<<(Md * DMODEL) / 256, 256, 0, stream>>>(dec_ids, tok_emb, pos_dec, xd, SEQ_D);
    for (int i = 0; i < 2; ++i) {
        ln(xd, dec_ln1_g + i * DMODEL, dec_ln1_b + i * DMODEL, hd_b, Md);
        gemm(hd_b, w_in_d + (size_t)i * 3 * DMODEL * DMODEL, dec_in_b + i * 3 * DMODEL,
             nullptr, qkvd, nullptr, Md, 3 * DMODEL, DMODEL, 0);
        k_attn_causal<<<(BATCH * NHEAD * SEQ_D) / 8, 256, 0, stream>>>(qkvd, aod_b);
        gemm(aod_b, w_sao_d + (size_t)i * DMODEL * DMODEL, dec_sao_b + i * DMODEL,
             xd, xd, nullptr, Md, DMODEL, DMODEL, 0);
        ln(xd, dec_ln2_g + i * DMODEL, dec_ln2_b + i * DMODEL, hd_b, Md);
        gemm(hd_b, w_q_d + (size_t)i * DMODEL * DMODEL, dec_q_b + i * DMODEL,
             nullptr, qc, nullptr, Md, DMODEL, DMODEL, 0);
        gemm(mem_b, w_kv_d + (size_t)i * 2 * DMODEL * DMODEL, dec_kv_b + i * 2 * DMODEL,
             nullptr, kvc, nullptr, Me, 2 * DMODEL, DMODEL, 0);
        k_attn_cross<<<(BATCH * NHEAD * SEQ_D) / 8, 256, 0, stream>>>(qc, kvc, attn_mask, aod_b);
        gemm(aod_b, w_co_d + (size_t)i * DMODEL * DMODEL, dec_co_b + i * DMODEL,
             xd, xd, nullptr, Md, DMODEL, DMODEL, 0);
        ln(xd, dec_ln3_g + i * DMODEL, dec_ln3_b + i * DMODEL, hd_b, Md);
        gemm(hd_b, w_fc1_d + (size_t)i * 4 * DMODEL * DMODEL, dec_fc1_b + i * 4 * DMODEL,
             nullptr, nullptr, ffnd_b, Md, 4 * DMODEL, DMODEL, 1);
        gemm(ffnd_b, w_fc2_d + (size_t)i * DMODEL * 4 * DMODEL, dec_fc2_b + i * DMODEL,
             xd, xd, nullptr, Md, DMODEL, 4 * DMODEL, 0);
    }
    ln(xd, ln_dec_g, ln_dec_b, hd_b, Md);

    // ================= LM head: [1024,512] x [32000,512]^T -> d_out =========
    gemm(hd_b, w_lm, nullptr, nullptr, (float*)d_out, nullptr, Md, 32000, DMODEL, 0);
}
